// Model_25623774888392
// MI455X (gfx1250) — matrix operand order repaired and checked by reading, not compiled, not run on hardware
//
#include <hip/hip_runtime.h>

// ---------------- types ----------------
typedef __attribute__((ext_vector_type(16))) __bf16 v16bf;
typedef __attribute__((ext_vector_type(8)))  float  v8f;

#define BATCH   64
#define NT      17472     // 64 + 64*16 + 64*256  (news list == user list length)
#define DIM     256

// f32 -> bf16 (round to nearest even), stored as u16 so host never touches __bf16
__device__ __forceinline__ unsigned short f2bfu(float f) {
    union { float f; unsigned u; } v; v.f = f;
    unsigned r = v.u + 0x7fffu + ((v.u >> 16) & 1u);
    return (unsigned short)(r >> 16);
}
__device__ __forceinline__ float wred(float v) {
    #pragma unroll
    for (int o = 16; o > 0; o >>= 1) v += __shfl_xor(v, o, 32);
    return v;
}

// Fragment-order maps used here (contiguous k, not the ISA table order; bf16 A 16x32 / B 32x16, wave32):
//  A: element e of lane l=(m|hf<<4) holds kk = e + 16*hf
//     -> writer of (m,kk): lane = m + 16*(kk>>4), e = kk&15
//  B: element e of lane l=(n|hf<<4) holds kk = e + 16*hf
//     -> writer of (kk,n): lane = n + 16*(kk>>4), e = kk&15

// ---------------- index prep ----------------
__global__ void k_prep(const int* __restrict__ user_idx, const int* __restrict__ news_idx,
                       const int* __restrict__ user_news, const int* __restrict__ news_user,
                       int* __restrict__ NewsList, int* __restrict__ UserList)
{
    int b = threadIdx.x;
    if (b >= BATCH) return;
    int n0 = news_idx[b], u0 = user_idx[b];
    NewsList[b] = n0;
    UserList[b] = u0;
    for (int j = 0; j < 16; ++j) {
        int n1 = news_user[n0 * 16 + j];   // user id
        int u1 = user_news[u0 * 16 + j];   // news id
        UserList[64 + b * 16 + j] = n1;
        NewsList[64 + b * 16 + j] = u1;
        for (int kk = 0; kk < 16; ++kk) {
            NewsList[1088 + (b * 16 + j) * 16 + kk] = user_news[n1 * 16 + kk];
            UserList[1088 + (b * 16 + j) * 16 + kk] = news_user[u1 * 16 + kk];
        }
    }
}

// ---------------- weight conversion into fragment-tile order ----------------
// dst panel layout: [kt][nt][lane][16] bf16, kt = Kpad/32 tiles, nt = N/16 tiles
__global__ void k_cvt_pad(unsigned short* __restrict__ dst, const float* __restrict__ src,
                          int K, int N, int Kpad)
{
    int i = blockIdx.x * blockDim.x + threadIdx.x;
    if (i >= Kpad * N) return;
    int k = i / N, n = i % N;
    float v = (k < K) ? src[(long)k * N + n] : 0.f;
    int kt = k >> 5, kk = k & 31, nt = n >> 4, nl = n & 15;
    int lane = nl + 16 * (kk >> 4), e = kk & 15;
    size_t di = (((size_t)kt * (N >> 4) + nt) * 32 + lane) * 16 + e;
    dst[di] = f2bfu(v);
}
__global__ void k_cvt_conv(unsigned short* __restrict__ dst, const float* __restrict__ ciw)
{
    int i = blockIdx.x * blockDim.x + threadIdx.x;   // 800*16
    if (i >= 800 * 16) return;
    int k = i / 16, c = i % 16;
    float v = (c < 8) ? ciw[c * 800 + k] : 0.f;      // ciw: (8,1,40,20) row-major
    int kt = k >> 5, kk = k & 31;
    int lane = c + 16 * (kk >> 4), e = kk & 15;
    dst[((size_t)kt * 32 + lane) * 16 + e] = f2bfu(v);
}

// ---------------- generic WMMA GEMM: C = op(A[idx] @ Bw + bias) ----------------
// A: f32 (optionally row-gathered); Bw: bf16 fragment panel [Kpad/32][N/16][32][16]
// Requires: M % 16 == 0, N % 16 == 0, K % 4 == 0, lda % 4 == 0, bias != null.
__global__ __launch_bounds__(256)
void k_gemm(const float* __restrict__ A, const int* __restrict__ aidx, int lda, int K,
            const unsigned short* __restrict__ Bw, int N, const float* __restrict__ bias,
            float* __restrict__ C, int ldc, int relu)
{
    __shared__ v16bf aFrag[32];
    int m0 = blockIdx.x * 16, n0 = blockIdx.y * 128;
    int t = threadIdx.x, w = t >> 5, l = t & 31;
    int mm = l & 15, hf = l >> 4;
    int ntiles = N >> 4;
    int nt = (n0 >> 4) + w;
    int active = (n0 + w * 16) < N;                  // wave-uniform
    const v16bf* Bf = (const v16bf*)Bw;

    // staging role (threads 0..127): owns row m=t>>3, k-quad kks=(t&7)*4; hoisted for all kt
    const float* arow = nullptr;
    unsigned* ap32 = (unsigned*)aFrag;
    int sidx = 0;
    if (t < 128) {
        int sm = t >> 3, kks = (t & 7) * 4;
        int row = aidx ? aidx[m0 + sm] : (m0 + sm);
        arow = A + (size_t)row * lda + kks;          // advance by 32 per k-tile
        int lane_s = sm + 16 * (kks >> 4);
        int e0 = kks & 15;
        sidx = (lane_s * 16 + e0) >> 1;              // dword index, even -> 8B aligned
    }

    v8f acc = {};
    int nk = (K + 31) / 32;
    for (int kt = 0; kt < nk; ++kt) {
        if (t < 128) {
            int k = kt * 32 + ((t & 7) * 4);
            float4 v = {0.f, 0.f, 0.f, 0.f};
            if (k + 3 < K) v = *(const float4*)(arow + (size_t)kt * 32);
            uint2 d;
            d.x = (unsigned)f2bfu(v.x) | ((unsigned)f2bfu(v.y) << 16);
            d.y = (unsigned)f2bfu(v.z) | ((unsigned)f2bfu(v.w) << 16);
            *(uint2*)&ap32[sidx] = d;                // ds_store_b64, fragment order
        }
        __syncthreads();
        if (active) {
            v16bf af = aFrag[l];                                        // 2x ds_load_b128
            v16bf bfr = Bf[((size_t)kt * ntiles + nt) * 32 + l];        // 2x global_load_b128
            if (kt + 1 < nk)                                            // prefetch next B tile
                __builtin_prefetch((const void*)&Bf[((size_t)(kt + 1) * ntiles + nt) * 32 + l], 0, 1);
            acc = __builtin_amdgcn_wmma_f32_16x16x32_bf16(false, af, false, bfr,
                                                          (short)0, acc, false, false);
        }
        __syncthreads();
    }
    if (active) {                                    // C: lane = N, vgpr r + 8*hf = M
        int nn = n0 + w * 16 + mm;
        float bv = bias[nn];
        float* cp = C + (size_t)(m0 + 8 * hf) * ldc + nn;
        #pragma unroll
        for (int r = 0; r < 8; ++r) {
            float v = acc[r] + bv;
            if (relu) v = fmaxf(v, 0.f);
            cp[(size_t)r * ldc] = v;
        }
    }
}

// ---------------- news convolution (item conv via WMMA + title conv + pooling) ----------------
__global__ __launch_bounds__(128)
void k_conv(const int* __restrict__ newsList,
            const int* __restrict__ news_title, const int* __restrict__ news_entity,
            const int* __restrict__ news_group,
            const float* __restrict__ word_emb, const float* __restrict__ group_emb,
            const unsigned short* __restrict__ wc,  // bf16 fragment panel [25][32][16]
            const float* __restrict__ cib, const float* __restrict__ ctw,
            const float* __restrict__ ctb, float* __restrict__ feat)
{
    __shared__ unsigned short ig[80 * 50];   // interleaved entity/group embeddings (bf16)
    __shared__ v16bf wl[25 * 32];            // conv weight fragments
    __shared__ float hi[21 * 16 * 8];        // item conv output
    __shared__ float te[10 * 50];            // title embeddings
    __shared__ float cw[8 * 40];             // title conv weights
    __shared__ float ht[8 * 5 * 16];         // title conv output

    int nb = blockIdx.x;
    int nid = newsList[nb];
    int t = threadIdx.x;

    {   // copy weight fragments (25.6 KB) as uint4
        uint4* d = (uint4*)wl; const uint4* s = (const uint4*)wc;
        for (int e = t; e < 1600; e += 128) d[e] = s[e];
    }
    for (int e = t; e < 4000; e += 128) {
        int row = e / 50, col = e % 50, j = row >> 1;
        float v;
        if (row & 1) { int gid = news_group[nid * 40 + j];  v = group_emb[(long)gid * 50 + col]; }
        else         { int wid = news_entity[nid * 40 + j]; v = word_emb[(long)wid * 50 + col]; }
        ig[e] = f2bfu(v);
    }
    for (int e = t; e < 500; e += 128) {
        int row = e / 50, col = e % 50;
        int wid = news_title[nid * 10 + row];
        te[e] = word_emb[(long)wid * 50 + col];
    }
    for (int e = t; e < 320; e += 128) cw[e] = ctw[e];
    __syncthreads();

    int w = t >> 5, l = t & 31;
    int mm = l & 15, hf = l >> 4;
    const unsigned short* igp = ig;
    for (int h = w; h < 21; h += 4) {                    // M-tile = output row h (16 cols)
        v8f acc = {};
        int abase = 100 * h + 2 * mm;                    // ig index base for this tile/lane
        #pragma unroll                                   // k constant -> /20,%20 fold to imm offsets
        for (int kt = 0; kt < 25; ++kt) {                // K = 800 = 25x32
            v16bf af;
            unsigned short* afp = (unsigned short*)&af;
            #pragma unroll
            for (int e = 0; e < 16; ++e) {
                int k = kt * 32 + e + 16 * hf;
                int r = k / 20, s = k % 20;
                afp[e] = igp[abase + r * 50 + s];
            }
            v16bf bfr = wl[kt * 32 + l];                 // 2x ds_load_b128
            acc = __builtin_amdgcn_wmma_f32_16x16x32_bf16(false, af, false, bfr,
                                                          (short)0, acc, false, false);
        }
        if (mm < 8) {                                    // lane = channel, vgpr = w-pos
            #pragma unroll
            for (int r = 0; r < 8; ++r) {
                int wp = r + 8 * hf;
                hi[(h * 16 + wp) * 8 + mm] = fmaxf(acc[r] + cib[mm], 0.f);
            }
        }
    }
    // title conv: 8ch x 5 x 16, kernel (2,20) stride (2,2)
    for (int e = t; e < 640; e += 128) {
        int c = e / 80, rem = e % 80, y = rem / 16, x = rem % 16;
        float s = ctb[c];
        #pragma unroll
        for (int r = 0; r < 2; ++r)
            for (int ss = 0; ss < 20; ++ss)
                s += te[(2 * y + r) * 50 + 2 * x + ss] * cw[c * 40 + r * 20 + ss];
        ht[(c * 5 + y) * 16 + x] = fmaxf(s, 0.f);
    }
    __syncthreads();
    // pooling + transpose -> feat[nb][896]
    float* out = feat + (long)nb * 896;
    for (int e = t; e < 640; e += 128) {                 // item pool (3,2)/(2,2) -> [10][8][8]
        int hp = e / 64, rem = e % 64, wp = rem / 8, c = rem % 8;
        float m = -1e30f;
        for (int r = 0; r < 3; ++r)
            for (int cc = 0; cc < 2; ++cc)
                m = fmaxf(m, hi[((2 * hp + r) * 16 + (2 * wp + cc)) * 8 + c]);
        out[e] = m;
    }
    for (int e = t; e < 256; e += 128) {                 // title pool (2,1)/(1,2) -> [4][8][8]
        int y = e / 64, rem = e % 64, x = rem / 8, c = rem % 8;
        float a = ht[(c * 5 + y) * 16 + 2 * x];
        float b = ht[(c * 5 + y + 1) * 16 + 2 * x];
        out[640 + e] = fmaxf(a, b);
    }
}

// ---------------- dynamic routing: 8 waves = 8 capsules, lane = 32-dim ----------------
__global__ __launch_bounds__(256)
void k_route(const float* __restrict__ selfv, const float* __restrict__ neighv,
             float* __restrict__ outp, int nnodes)
{
    __shared__ float lgt[16 * 8];
    int node = blockIdx.x;
    if (node >= nnodes) return;
    int t = threadIdx.x, k = t >> 5, d = t & 31;
    float x = selfv[(long)node * DIM + k * 32 + d];
    x *= rsqrtf(wred(x * x) + 1e-12f);
    float z[16];
    #pragma unroll
    for (int m = 0; m < 16; ++m) {
        float v = neighv[((long)node * 16 + m) * DIM + k * 32 + d];
        z[m] = v * rsqrtf(wred(v * v) + 1e-12f);
    }
    float u = x;
    for (int it = 0; it < 7; ++it) {
        float lg[16];
        #pragma unroll
        for (int m = 0; m < 16; ++m) lg[m] = wred(u * z[m]);   // all lanes hold value
        __syncthreads();
        if (d < 16) lgt[d * 8 + k] = lg[d];
        __syncthreads();
        float acc = 0.f;
        #pragma unroll
        for (int m = 0; m < 16; ++m) {                          // softmax over capsules k
            float mx = -1e30f;
            #pragma unroll
            for (int kk = 0; kk < 8; ++kk) mx = fmaxf(mx, lgt[m * 8 + kk]);
            float den = 0.f;
            #pragma unroll
            for (int kk = 0; kk < 8; ++kk) den += __expf(lgt[m * 8 + kk] - mx);
            acc += (__expf(lg[m] - mx) / den) * z[m];
        }
        float un = x + acc;
        u = un * rsqrtf(wred(un * un) + 1e-12f);
    }
    outp[(long)node * DIM + k * 32 + d] = u;
}

// ---------------- host ----------------
extern "C" void kernel_launch(void* const* d_in, const int* in_sizes, int n_in,
                              void* d_out, int out_size, void* d_ws, size_t ws_size,
                              hipStream_t stream) {
    (void)in_sizes; (void)n_in; (void)out_size; (void)ws_size;
    const int*   user_indices = (const int*)d_in[0];
    const int*   news_indices = (const int*)d_in[1];
    const int*   user_news    = (const int*)d_in[2];
    const int*   news_user    = (const int*)d_in[3];
    const int*   news_title   = (const int*)d_in[4];
    const int*   news_entity  = (const int*)d_in[5];
    const int*   news_group   = (const int*)d_in[6];
    const float* word_emb     = (const float*)d_in[7];
    const float* group_emb    = (const float*)d_in[8];
    const float* user_emb     = (const float*)d_in[9];
    const float* ciw          = (const float*)d_in[10];
    const float* cib          = (const float*)d_in[11];
    const float* ctw          = (const float*)d_in[12];
    const float* ctb          = (const float*)d_in[13];
    const float* dw           = (const float*)d_in[14];
    const float* db           = (const float*)d_in[15];
    const float* utw          = (const float*)d_in[16];
    const float* utb          = (const float*)d_in[17];
    const float* itw          = (const float*)d_in[18];
    const float* itb          = (const float*)d_in[19];
    const float* rw           = (const float*)d_in[20];
    const float* rb           = (const float*)d_in[21];
    float* out = (float*)d_out;

    size_t off = 0;
    auto alloc = [&](size_t bytes) -> void* {
        void* p = (char*)d_ws + off;
        off += (bytes + 255) & ~(size_t)255;
        return p;
    };
    int* NewsList = (int*)alloc(NT * sizeof(int));
    int* UserList = (int*)alloc(NT * sizeof(int));
    unsigned short* bWu = (unsigned short*)alloc((size_t)128 * 256 * 2);
    unsigned short* bWd = (unsigned short*)alloc((size_t)896 * 400 * 2);
    unsigned short* bWi = (unsigned short*)alloc((size_t)416 * 256 * 2);  // K padded 400->416
    unsigned short* bWr = (unsigned short*)alloc((size_t)256 * 256 * 2);
    unsigned short* bWc = (unsigned short*)alloc((size_t)800 * 16 * 2);
    float* uv   = (float*)alloc((size_t)NT * 256 * 4);
    float* feat = (float*)alloc((size_t)NT * 896 * 4);
    float* dn   = (float*)alloc((size_t)NT * 400 * 4);
    float* nv   = feat;                                   // reuse: feat dead after dense GEMM
    float* H    = (float*)alloc((size_t)2176 * 256 * 4);
    float* T    = (float*)alloc((size_t)2176 * 256 * 4);

    // 1) index lists
    k_prep<<<1, 64, 0, stream>>>(user_indices, news_indices, user_news, news_user,
                                 NewsList, UserList);
    // 2) weight conversion to bf16 fragment panels
    k_cvt_pad<<<(128 * 256 + 255) / 256, 256, 0, stream>>>(bWu, utw, 128, 256, 128);
    k_cvt_pad<<<(896 * 400 + 255) / 256, 256, 0, stream>>>(bWd, dw, 896, 400, 896);
    k_cvt_pad<<<(416 * 256 + 255) / 256, 256, 0, stream>>>(bWi, itw, 400, 256, 416);
    k_cvt_pad<<<(256 * 256 + 255) / 256, 256, 0, stream>>>(bWr, rw, 256, 256, 256);
    k_cvt_conv<<<(800 * 16 + 255) / 256, 256, 0, stream>>>(bWc, ciw);
    // 3) user transform: uv = relu(user_emb[UserList] @ Wu + bu)
    k_gemm<<<dim3(NT / 16, 2), 256, 0, stream>>>(user_emb, UserList, 128, 128,
                                                 bWu, 256, utb, uv, 256, 1);
    // 4) news convolutions -> feat[NT][896]
    k_conv<<<NT, 128, 0, stream>>>(NewsList, news_title, news_entity, news_group,
                                   word_emb, group_emb, bWc, cib, ctw, ctb, feat);
    // 5) dense: dn = feat @ Wd + db (no relu)
    k_gemm<<<dim3(NT / 16, 4), 256, 0, stream>>>(feat, nullptr, 896, 896,
                                                 bWd, 400, db, dn, 400, 0);
    // 6) item transform: nv = relu(dn @ Wi + bi)
    k_gemm<<<dim3(NT / 16, 2), 256, 0, stream>>>(dn, nullptr, 400, 400,
                                                 bWi, 256, itb, nv, 256, 1);
    // 7) routing level 0/1  (H rows: [0:64)=n_h0, [64:1088)=n_h1, [1088:1152)=u_h0, [1152:2176)=u_h1)
    k_route<<<64,   256, 0, stream>>>(nv,             uv + 64 * 256,   H,               64);
    k_route<<<1024, 256, 0, stream>>>(uv + 64 * 256,  nv + 1088 * 256, H + 64 * 256,    1024);
    k_route<<<64,   256, 0, stream>>>(uv,             nv + 64 * 256,   H + 1088 * 256,  64);
    k_route<<<1024, 256, 0, stream>>>(nv + 64 * 256,  uv + 1088 * 256, H + 1152 * 256,  1024);
    // 8) router1 transform: T = relu(H @ Wr + br)
    k_gemm<<<dim3(2176 / 16, 2), 256, 0, stream>>>(H, nullptr, 256, 256,
                                                   bWr, 256, rb, T, 256, 1);
    // 9) final routes -> output (u_out first, then n_out)
    k_route<<<64, 256, 0, stream>>>(T + 1088 * 256, T + 1152 * 256, out,         64);
    k_route<<<64, 256, 0, stream>>>(T,              T + 64 * 256,   out + 16384, 64);
}
